// STBlock_39264591020395
// MI455X (gfx1250) — matrix operand order repaired and checked by reading, not compiled, not run on hardware
//
#include <hip/hip_runtime.h>
#include <stdint.h>

// ---------------------------------------------------------------------------
// ST-GCN block for MI455X (gfx1250, wave32). All heavy math is routed through
// v_wmma_f32_16x16x32_bf16 (bf16 inputs, f32 accumulate). ~210 GFLOP of
// N x N graph GEMMs + ~11 GFLOP of thin K=64 GEMMs (Theta / 1x3 conv / 1x1
// residual), all expressed as WMMA GEMMs. GEMM tiles are staged with
// double-buffered GLOBAL_LOAD_ASYNC_TO_LDS_B128 (ASYNCcnt) overlapped with
// the WMMA issue. Elementwise / softmax / LN are VALU. Workspace ~140 MB.
// ---------------------------------------------------------------------------

typedef __bf16 bf16_t;
typedef __attribute__((ext_vector_type(16))) __bf16 v16bf;
typedef __attribute__((ext_vector_type(8)))  float  v8f;

#define BB  8
#define NN_ 2048
#define CC  64
#define TT  12
#define KK  3
#define OO  64

#define USE_ASYNC_LDS 1

// ---- CDNA5 async global->LDS copy (CDNA5 ISA) ----
// Per-lane: LDS[vdst32] = MEM[vaddr64], 16 bytes. Tracked by ASYNCcnt.
// Generic pointers to __shared__ carry the LDS byte offset in their low
// 32 bits (ISA aperture rule), which is exactly what VDST wants.
__device__ __forceinline__ void async_ld16(uint32_t lds_byte_addr, const void* gaddr) {
  asm volatile("global_load_async_to_lds_b128 %0, %1, off"
               :: "v"(lds_byte_addr), "v"(gaddr) : "memory");
}
__device__ __forceinline__ void s_wait_async_le4() {
  asm volatile("s_wait_asynccnt 0x4" ::: "memory");  // retire older tile only
}
__device__ __forceinline__ void s_wait_async_le0() {
  asm volatile("s_wait_asynccnt 0x0" ::: "memory");
}

// ========================= generic WMMA NT GEMM ============================
// C(MxN) = A(MxK) @ Bt(NxK)^T, A/Bt bf16 row-major, f32 accumulate.
// out_mode: 0 = f32 store, 1 = f32 +=, 2 = f32 += with conv-pad row remap
//           (row -> (row/12)*14 + row%12 + 1), 3 = bf16 transposed store
//           (Cb[n*ldcb+m]), 4 = bf16 store (Cb[m*ldcb+n]).
// a_remap: 0 direct, 1 row -> (row/12)*14 + row%12 + a_shift (conv gather).
// z-batch: A += z*az, Bt += z*bz, C += z*cz (elements of its own type).

__device__ __forceinline__ v16bf lds_load_afrag(const bf16_t* As, int stride,
                                                int base_row, int lane) {
  // 16-bit A 16x32 layout: lane m (0..15): K 0..7 then 16..23;
  // lane m+16: K 8..15 then 24..31 (ISA 7.12.2)
  int row = base_row + (lane & 15);
  int k0  = (lane < 16) ? 0 : 8;
  const bf16_t* p = As + row * stride + k0;
  v16bf a;
  ((uint4*)&a)[0] = *(const uint4*)(p);
  ((uint4*)&a)[1] = *(const uint4*)(p + 16);
  return a;
}

__device__ __forceinline__ v16bf lds_load_bfrag(const bf16_t* Bs, int stride,
                                                int base_col, int lane) {
  // 16-bit B 32x16 layout: lane n (0..15): K 0..7 then 16..23; lane n+16: K 8..15 then 24..31
  int col = base_col + (lane & 15);
  int k0  = (lane < 16) ? 0 : 8;
  const bf16_t* p = Bs + col * stride + k0;
  v16bf b;
  ((uint4*)&b)[0] = *(const uint4*)(p);
  ((uint4*)&b)[1] = *(const uint4*)(p + 16);
  return b;
}

__global__ __launch_bounds__(256) void k_gemm_nt(
    const bf16_t* __restrict__ A, int lda, int a_remap, int a_shift, long long az,
    const bf16_t* __restrict__ Bt, int ldb, long long bz,
    float* __restrict__ Cf, int ldc, long long cz,
    bf16_t* __restrict__ Cb, int ldcb,
    int out_mode, int M, int N, int K)
{
  constexpr int TM = 128, TN = 128, TK = 32, LSTR = TK + 8;
  __shared__ bf16_t As[2][TM * LSTR];
  __shared__ bf16_t Bs[2][TN * LSTR];

  const int tid  = threadIdx.x;
  const int lane = tid & 31;
  const int wave = tid >> 5;
  const int wm = wave >> 1, wn = wave & 1;       // 4 x 2 wave grid
  const int bm = blockIdx.y * TM, bn = blockIdx.x * TN;

  A  += (long long)blockIdx.z * az;
  Bt += (long long)blockIdx.z * bz;
  if (Cf) Cf += (long long)blockIdx.z * cz;
  if (Cb) Cb += (long long)blockIdx.z * cz;

  v8f acc[2][4];
  const v8f vzero = {0.f,0.f,0.f,0.f,0.f,0.f,0.f,0.f};
  for (int i = 0; i < 2; ++i)
    for (int j = 0; j < 4; ++j) acc[i][j] = vzero;

  // Cooperative tile staging: 2 threads per row, 16 halves (2 x b128) each.
  // Out-of-range rows are CLAMPED to a valid address instead of masked:
  // the garbage only feeds output rows/cols the epilogue never stores, and
  // every wave then issues exactly 4 async ops per k-step (boundaries are
  // multiples of 16 rows = one wave's staging group), keeping the ASYNCcnt
  // thresholds uniform.
  const int lrow = tid >> 1;
  const int lseg = (tid & 1) * 16;
  int gra = bm + lrow; if (gra > M - 1) gra = M - 1;
  const long long arow = a_remap
      ? (long long)((gra / 12) * 14 + gra % 12 + a_shift)
      : (long long)gra;
  int gcb = bn + lrow; if (gcb > N - 1) gcb = N - 1;

  const bf16_t* Ap = A + arow * lda + lseg;
  const bf16_t* Bp = Bt + (long long)gcb * ldb + lseg;

  uint32_t aoff[2], boff[2];
  aoff[0] = (uint32_t)(uintptr_t)&As[0][lrow * LSTR + lseg];
  aoff[1] = (uint32_t)(uintptr_t)&As[1][lrow * LSTR + lseg];
  boff[0] = (uint32_t)(uintptr_t)&Bs[0][lrow * LSTR + lseg];
  boff[1] = (uint32_t)(uintptr_t)&Bs[1][lrow * LSTR + lseg];

#if USE_ASYNC_LDS
  // prologue: stage k-tile 0 into buffer 0
  async_ld16(aoff[0],      Ap);
  async_ld16(aoff[0] + 16, Ap + 8);
  async_ld16(boff[0],      Bp);
  async_ld16(boff[0] + 16, Bp + 8);
#endif

  int cur = 0;
  for (int kb = 0; kb < K; kb += TK) {
#if USE_ASYNC_LDS
    if (kb + TK < K) {
      const int nb = cur ^ 1;
      const bf16_t* An = Ap + kb + TK;
      const bf16_t* Bn = Bp + kb + TK;
      async_ld16(aoff[nb],      An);
      async_ld16(aoff[nb] + 16, An + 8);
      async_ld16(boff[nb],      Bn);
      async_ld16(boff[nb] + 16, Bn + 8);
      s_wait_async_le4();   // async loads retire in order: current tile done
    } else {
      s_wait_async_le0();
    }
#else
    {
      const uint4* pa = (const uint4*)(Ap + kb);
      const uint4* pb = (const uint4*)(Bp + kb);
      uint4 a0 = pa[0], a1 = pa[1], b0 = pb[0], b1 = pb[1];
      *(uint4*)((bf16_t*)As[cur] + lrow * LSTR + lseg)     = a0;
      *(uint4*)((bf16_t*)As[cur] + lrow * LSTR + lseg + 8) = a1;
      *(uint4*)((bf16_t*)Bs[cur] + lrow * LSTR + lseg)     = b0;
      *(uint4*)((bf16_t*)Bs[cur] + lrow * LSTR + lseg + 8) = b1;
    }
#endif
    __syncthreads();

    const bf16_t* Asb = &As[cur][0];
    const bf16_t* Bsb = &Bs[cur][0];
    v16bf af[2], bfr[4];
    af[0] = lds_load_afrag(Asb, LSTR, wm * 32,      lane);
    af[1] = lds_load_afrag(Asb, LSTR, wm * 32 + 16, lane);
    for (int nt = 0; nt < 4; ++nt)
      bfr[nt] = lds_load_bfrag(Bsb, LSTR, wn * 64 + nt * 16, lane);

    for (int mt = 0; mt < 2; ++mt)
      for (int nt = 0; nt < 4; ++nt)
        acc[mt][nt] = __builtin_amdgcn_wmma_f32_16x16x32_bf16(
            false, af[mt], false, bfr[nt], (short)0, acc[mt][nt], false, false);
    __syncthreads();
    cur ^= 1;
  }

  // Epilogue. D layout: VGPR v, lanes 0-15 -> M=v, lanes 16-31 -> M=8+v, N=lane&15.
  const int mlo = (lane >= 16) ? 8 : 0;
  for (int mt = 0; mt < 2; ++mt) {
    for (int nt = 0; nt < 4; ++nt) {
      const int ng = bn + wn * 64 + nt * 16 + (lane & 15);
      if (ng >= N) continue;
      const int mbase = bm + wm * 32 + mt * 16 + mlo;
      v8f a = acc[mt][nt];
      for (int v = 0; v < 8; ++v) {
        const int mg = mbase + v;
        if (mg >= M) continue;
        const float val = a[v];
        switch (out_mode) {
          case 0: Cf[(long long)mg * ldc + ng] = val; break;
          case 1: Cf[(long long)mg * ldc + ng] += val; break;
          case 2: {
            const int cr = (mg / 12) * 14 + mg % 12 + 1;
            Cf[(long long)cr * ldc + ng] += val;
          } break;
          case 3: Cb[(long long)ng * ldcb + mg] = (bf16_t)val; break;
          case 4: Cb[(long long)mg * ldcb + ng] = (bf16_t)val; break;
        }
      }
    }
  }
}

// ============================ small kernels ================================

// weights prep: W13 = W1@W3 transposed to bf16, W2^T, Theta^T, tconv^T, res_w
__global__ void k_prep(const float* __restrict__ W1, const float* __restrict__ W2,
                       const float* __restrict__ W3, const float* __restrict__ Th,
                       const float* __restrict__ tw, const float* __restrict__ rw,
                       bf16_t* __restrict__ W13T, bf16_t* __restrict__ W2T,
                       bf16_t* __restrict__ ThT,  bf16_t* __restrict__ twT,
                       bf16_t* __restrict__ rwT)
{
  __shared__ float w13[64 * 64];
  const int tid = threadIdx.x;
  for (int i = tid; i < 4096; i += 256) {
    int c = i >> 6, j = i & 63;
    float s = 0.f;
    for (int q = 0; q < 64; ++q) s += W1[c * 64 + q] * W3[q * 64 + j];
    w13[i] = s;
  }
  __syncthreads();
  for (int i = tid; i < 4096; i += 256) {
    int j = i >> 6, c = i & 63;
    W13T[i] = (bf16_t)w13[c * 64 + j];
    W2T[i]  = (bf16_t)W2[c * 64 + j];
    rwT[i]  = (bf16_t)rw[i];                       // res_w already [o][c]
  }
  for (int i = tid; i < 3 * 4096; i += 256) {
    int k = i >> 12, r = i & 4095, o = r >> 6, c = r & 63;
    ThT[i] = (bf16_t)Th[k * 4096 + c * 64 + o];    // Theta[k,c,o] -> [k][o][c]
    twT[i] = (bf16_t)tw[o * 192 + c * 3 + k];      // tconv[o,o',0,dt] -> [dt][o][o']
  }
}

__global__ void k_cast_bf16(const float* __restrict__ s, bf16_t* __restrict__ d, int n) {
  int i = blockIdx.x * 256 + threadIdx.x;
  if (i < n) d[i] = (bf16_t)s[i];
}

// x (B,N,C,T) f32 -> xT[b][(n*12+t)*64 + c] bf16  (rows over (n,t), cols c)
__global__ void k_xpose(const float* __restrict__ x, bf16_t* __restrict__ xT) {
  int i = blockIdx.x * 256 + threadIdx.x;
  if (i >= BB * NN_ * TT * CC) return;
  int c = i & 63;
  int r = i >> 6;            // (b*N + n)*T + t
  int t = r % TT;
  long long bn = r / TT;     // b*N + n
  xT[i] = (bf16_t)x[(bn * CC + c) * TT + t];
}

// Xt[b,t,c] = sum_n x[b,n,c,t]
__global__ void k_xtsum(const float* __restrict__ x, float* __restrict__ Xt) {
  int b = blockIdx.x >> 6, c = blockIdx.x & 63;
  int tid = threadIdx.x;
  float acc[TT];
  for (int t = 0; t < TT; ++t) acc[t] = 0.f;
  for (int n = tid; n < NN_; n += 256) {
    const float* p = x + (((long long)b * NN_ + n) * CC + c) * TT;
    for (int t = 0; t < TT; ++t) acc[t] += p[t];
  }
  __shared__ float red[256];
  for (int t = 0; t < TT; ++t) {
    red[tid] = acc[t];
    __syncthreads();
    for (int s = 128; s > 0; s >>= 1) {
      if (tid < s) red[tid] += red[tid + s];
      __syncthreads();
    }
    if (tid == 0) Xt[(b * TT + t) * CC + c] = red[0];
    __syncthreads();
  }
}

// temporal attention matrix E (B,T,T); tiny, one block per batch
__global__ void k_attnE(const float* __restrict__ Xt, const float* __restrict__ U1,
                        const float* __restrict__ U2, const float* __restrict__ Ve,
                        const float* __restrict__ be, float* __restrict__ E)
{
  int b = blockIdx.x, tid = threadIdx.x;
  __shared__ float xs[TT * CC], t1[TT * CC], t2[TT * CC], P[TT * TT];
  for (int i = tid; i < TT * CC; i += 256) xs[i] = Xt[b * TT * CC + i];
  __syncthreads();
  for (int i = tid; i < TT * CC; i += 256) {
    int t = i >> 6, c = i & 63;
    float s1 = 0.f, s2 = 0.f;
    for (int q = 0; q < CC; ++q) {
      float xv = xs[t * CC + q];
      s1 += xv * U1[q * CC + c];
      s2 += xv * U2[q * CC + c];
    }
    t1[i] = s1; t2[i] = s2;
  }
  __syncthreads();
  const float be0 = be[0];
  for (int i = tid; i < TT * TT; i += 256) {
    int t = i / TT, u = i % TT;
    float s = 0.f;
    for (int q = 0; q < CC; ++q) s += t1[t * CC + q] * t2[u * CC + q];
    P[i] = tanhf(s + be0) * Ve[i];
  }
  __syncthreads();
  if (tid < TT) {
    int t = tid;
    float m = -1e30f;
    for (int u = 0; u < TT; ++u) m = fmaxf(m, P[t * TT + u]);
    float e[TT], s = 0.f;
    for (int u = 0; u < TT; ++u) { e[u] = __expf(P[t * TT + u] - m); s += e[u]; }
    float r = 1.f / s;
    for (int u = 0; u < TT; ++u) E[b * TT * TT + t * TT + u] = e[u] * r;
  }
}

// x_ta (temporal-attended) -> transposed bf16 [(u*64+c), n] per batch, and
// Xs[b,n,c] = sum_u x_ta (bf16, GEMM-A layout)
__global__ void k_temporal(const float* __restrict__ x, const float* __restrict__ E,
                           bf16_t* __restrict__ xtaT, bf16_t* __restrict__ Xs)
{
  int b = blockIdx.x >> 6, c = blockIdx.x & 63;
  int n = blockIdx.y * 256 + threadIdx.x;
  __shared__ float Es[TT * TT];
  if (threadIdx.x < TT * TT) Es[threadIdx.x] = E[b * TT * TT + threadIdx.x];
  __syncthreads();
  const float* p = x + (((long long)b * NN_ + n) * CC + c) * TT;
  float xr[TT];
  for (int t = 0; t < TT; ++t) xr[t] = p[t];
  float tot = 0.f;
  for (int u = 0; u < TT; ++u) {
    float a = 0.f;
    for (int t = 0; t < TT; ++t) a += xr[t] * Es[t * TT + u];
    xtaT[(long long)b * (TT * CC * NN_) + (u * CC + c) * NN_ + n] = (bf16_t)a;
    tot += a;
  }
  Xs[((long long)b * NN_ + n) * CC + c] = (bf16_t)tot;
}

// row softmax of tanh(Spre + bs)*Vs ; rewrites Spre in place, emits bf16 S
__global__ void k_softmaxS(float* __restrict__ Sp, const float* __restrict__ Vs,
                           const float* __restrict__ bs, bf16_t* __restrict__ Sb)
{
  int n = blockIdx.x, tid = threadIdx.x;
  float* row = Sp + (long long)n * NN_;
  const float* vr = Vs + (long long)n * NN_;
  __shared__ float red[256];
  const float bs0 = bs[0];
  float lmax = -1e30f;
  for (int m = tid; m < NN_; m += 256) {
    float v = tanhf(row[m] + bs0) * vr[m];
    row[m] = v;
    lmax = fmaxf(lmax, v);
  }
  red[tid] = lmax; __syncthreads();
  for (int s = 128; s > 0; s >>= 1) {
    if (tid < s) red[tid] = fmaxf(red[tid], red[tid + s]);
    __syncthreads();
  }
  const float mx = red[0];
  __syncthreads();
  float ls = 0.f;
  for (int m = tid; m < NN_; m += 256) {
    float e = __expf(row[m] - mx);
    row[m] = e;
    ls += e;
  }
  red[tid] = ls; __syncthreads();
  for (int s = 128; s > 0; s >>= 1) {
    if (tid < s) red[tid] += red[tid + s];
    __syncthreads();
  }
  const float rs = 1.f / red[0];
  for (int m = tid; m < NN_; m += 256)
    Sb[(long long)n * NN_ + m] = (bf16_t)(row[m] * rs);
}

// Gpad init: rows n*14 + (1..12) = gbias[o], pad rows (0,13) = 0
__global__ void k_init_gpad(const float* __restrict__ gbias, float* __restrict__ Gf) {
  int i = blockIdx.x * 256 + threadIdx.x;
  if (i >= 28672 * 64) return;
  int o = i & 63, row = i >> 6, r = row % 14;
  Gf[i] = (r >= 1 && r <= 12) ? gbias[o] : 0.f;
}

__global__ void k_init_out(const float* __restrict__ tb, const float* __restrict__ rb,
                           float* __restrict__ O) {
  int i = blockIdx.x * 256 + threadIdx.x;
  if (i >= 24576 * 64) return;
  O[i] = tb[i & 63] + rb[i & 63];
}

// LN stats per t over (N,O) of relu(OUT)
__global__ void k_stats(const float* __restrict__ O, float* __restrict__ st) {
  int t = blockIdx.x, tid = threadIdx.x;
  float s1 = 0.f, s2 = 0.f;
  for (int e = tid; e < NN_ * OO; e += 256) {
    int n = e >> 6, o = e & 63;
    float v = fmaxf(O[((long long)n * TT + t) * OO + o], 0.f);
    s1 += v; s2 += v * v;
  }
  __shared__ float r1[256], r2[256];
  r1[tid] = s1; r2[tid] = s2; __syncthreads();
  for (int s = 128; s > 0; s >>= 1) {
    if (tid < s) { r1[tid] += r1[tid + s]; r2[tid] += r2[tid + s]; }
    __syncthreads();
  }
  if (tid == 0) {
    const float inv = 1.f / (float)(NN_ * OO);
    float mu  = r1[0] * inv;
    float var = r2[0] * inv - mu * mu;
    st[t * 2]     = mu;
    st[t * 2 + 1] = rsqrtf(var + 1e-5f);
  }
}

// relu + LN + transpose to (n,o,t) per batch
__global__ void k_final(const float* __restrict__ O, const float* __restrict__ st,
                        const float* __restrict__ lw, const float* __restrict__ lb,
                        float* __restrict__ out) {
  int i = blockIdx.x * 256 + threadIdx.x;
  if (i >= NN_ * OO * TT) return;
  int t = i % TT, r = i / TT;
  int o = r % OO, n = r / OO;
  float v = fmaxf(O[((long long)n * TT + t) * OO + o], 0.f);
  out[i] = (v - st[t * 2]) * st[t * 2 + 1] * lw[n * OO + o] + lb[n * OO + o];
}

// ============================== host side ==================================

static void launch_gemm(hipStream_t s,
    const bf16_t* A, int lda, int arem, int ash, long long az,
    const bf16_t* Bt, int ldb, long long bz,
    float* Cf, int ldc, long long cz, bf16_t* Cb, int ldcb,
    int mode, int M, int N, int K, int Z)
{
  dim3 grid((N + 127) / 128, (M + 127) / 128, Z);
  k_gemm_nt<<<grid, 256, 0, s>>>(A, lda, arem, ash, az, Bt, ldb, bz,
                                 Cf, ldc, cz, Cb, ldcb, mode, M, N, K);
}

extern "C" void kernel_launch(void* const* d_in, const int* in_sizes, int n_in,
                              void* d_out, int out_size, void* d_ws, size_t ws_size,
                              hipStream_t stream)
{
  (void)in_sizes; (void)n_in; (void)out_size; (void)ws_size;
  const float* x     = (const float*)d_in[0];
  const float* cheb  = (const float*)d_in[1];
  const float* W1    = (const float*)d_in[2];
  const float* W2    = (const float*)d_in[3];
  const float* W3    = (const float*)d_in[4];
  const float* Vs    = (const float*)d_in[5];
  const float* bs    = (const float*)d_in[6];
  const float* U1    = (const float*)d_in[7];
  const float* U2    = (const float*)d_in[8];
  /* d_in[9] = U3 : defined but unused in the reference forward */
  const float* Ve    = (const float*)d_in[10];
  const float* be    = (const float*)d_in[11];
  const float* Th    = (const float*)d_in[12];
  const float* gbias = (const float*)d_in[13];
  const float* tw    = (const float*)d_in[14];
  const float* tb    = (const float*)d_in[15];
  const float* rw    = (const float*)d_in[16];
  const float* rb    = (const float*)d_in[17];
  const float* lnw   = (const float*)d_in[18];
  const float* lnb   = (const float*)d_in[19];
  float* out = (float*)d_out;

  char* w = (char*)d_ws;
  size_t off = 0;
  auto alloc = [&](size_t bytes) -> void* {
    void* p = w + off;
    off = (off + bytes + 255) & ~(size_t)255;
    return p;
  };
  float*  Xt    = (float*) alloc((size_t)BB * TT * CC * 4);
  float*  E     = (float*) alloc((size_t)BB * TT * TT * 4);
  bf16_t* W13T  = (bf16_t*)alloc(4096 * 2);
  bf16_t* W2T   = (bf16_t*)alloc(4096 * 2);
  bf16_t* ThT   = (bf16_t*)alloc(3 * 4096 * 2);
  bf16_t* twT   = (bf16_t*)alloc(3 * 4096 * 2);
  bf16_t* rwT   = (bf16_t*)alloc(4096 * 2);
  bf16_t* Xs    = (bf16_t*)alloc((size_t)BB * NN_ * CC * 2);
  bf16_t* f1    = (bf16_t*)alloc((size_t)BB * NN_ * CC * 2);
  bf16_t* f2    = (bf16_t*)alloc((size_t)BB * NN_ * CC * 2);
  bf16_t* chebb = (bf16_t*)alloc((size_t)KK * NN_ * NN_ * 2);
  bf16_t* xtaT  = (bf16_t*)alloc((size_t)BB * TT * CC * NN_ * 2);
  bf16_t* xT    = (bf16_t*)alloc((size_t)BB * NN_ * TT * CC * 2);
  float*  Spre  = (float*) alloc((size_t)NN_ * NN_ * 4);        // per-batch reuse
  bf16_t* Sbf   = (bf16_t*)alloc((size_t)NN_ * NN_ * 2);
  bf16_t* xsT   = (bf16_t*)alloc((size_t)TT * CC * NN_ * 2);
  bf16_t* supb  = (bf16_t*)alloc((size_t)KK * NN_ * TT * CC * 2);
  float*  Gf    = (float*) alloc((size_t)28672 * 64 * 4);
  bf16_t* Gb    = (bf16_t*)alloc((size_t)28672 * 64 * 2);
  float*  OUTb  = (float*) alloc((size_t)24576 * 64 * 4);
  float*  stats = (float*) alloc(24 * 4);

  // ---- setup / temporal attention (cheap, memory-bound) ----
  k_prep<<<1, 256, 0, stream>>>(W1, W2, W3, Th, tw, rw, W13T, W2T, ThT, twT, rwT);
  { int n = KK * NN_ * NN_;       k_cast_bf16<<<(n + 255) / 256, 256, 0, stream>>>(cheb, chebb, n); }
  { int n = BB * NN_ * TT * CC;   k_xpose<<<(n + 255) / 256, 256, 0, stream>>>(x, xT); }
  k_xtsum<<<BB * CC, 256, 0, stream>>>(x, Xt);
  k_attnE<<<BB, 256, 0, stream>>>(Xt, U1, U2, Ve, be, E);
  k_temporal<<<dim3(BB * CC, NN_ / 256), 256, 0, stream>>>(x, E, xtaT, Xs);

  // ---- spatial-attention projections: f1 = Xs@(W1W3), f2 = Xs@W2 (WMMA) ----
  launch_gemm(stream, Xs, 64, 0, 0, 0, W13T, 64, 0, nullptr, 0, 0, f1, 64, 4,
              BB * NN_, 64, 64, 1);
  launch_gemm(stream, Xs, 64, 0, 0, 0, W2T, 64, 0, nullptr, 0, 0, f2, 64, 4,
              BB * NN_, 64, 64, 1);

  for (int b = 0; b < BB; ++b) {
    const bf16_t* f1b = f1 + (size_t)b * NN_ * CC;
    const bf16_t* f2b = f2 + (size_t)b * NN_ * CC;
    // S_pre = f1W3 @ f2^T  (2048 x 2048, K=64)
    launch_gemm(stream, f1b, 64, 0, 0, 0, f2b, 64, 0, Spre, NN_, 0, nullptr, 0,
                0, NN_, NN_, 64, 1);
    // S = softmax(tanh(S_pre + bs) * Vs) -> bf16
    k_softmaxS<<<NN_, 256, 0, stream>>>(Spre, Vs, bs, Sbf);
    // x_s^T = (S @ x_ta)^T  (M=2048, N=768, K=2048) -> bf16 transposed
    launch_gemm(stream, Sbf, NN_, 0, 0, 0, xtaT + (size_t)b * TT * CC * NN_, NN_, 0,
                nullptr, 0, 0, xsT, NN_, 3, NN_, TT * CC, NN_, 1);
    // sup_k = cheb_k @ x_s  for k=0..2 (z-batched) -> bf16 direct
    launch_gemm(stream, chebb, NN_, 0, 0, (long long)NN_ * NN_, xsT, NN_, 0,
                nullptr, 0, (long long)NN_ * TT * CC, supb, TT * CC, 4,
                NN_, TT * CC, NN_, KK);
    // G[n,t,o] = gbias + sum_k sup_k @ Theta_k  (rows remapped into 14-row pad)
    k_init_gpad<<<(28672 * 64 + 255) / 256, 256, 0, stream>>>(gbias, Gf);
    for (int k = 0; k < KK; ++k)
      launch_gemm(stream, supb + (size_t)k * NN_ * TT * CC, 64, 0, 0, 0,
                  ThT + (size_t)k * 4096, 64, 0, Gf, 64, 0, nullptr, 0, 2,
                  NN_ * TT, 64, 64, 1);
    { int n = 28672 * 64; k_cast_bf16<<<(n + 255) / 256, 256, 0, stream>>>(Gf, Gb, n); }
    // OUT = tconv_b + res_b + x @ res_w^T + sum_dt shift(G, dt) @ w_dt^T
    k_init_out<<<(24576 * 64 + 255) / 256, 256, 0, stream>>>(tb, rb, OUTb);
    launch_gemm(stream, xT + (size_t)b * NN_ * TT * CC, 64, 0, 0, 0, rwT, 64, 0,
                OUTb, 64, 0, nullptr, 0, 1, NN_ * TT, 64, 64, 1);
    for (int dt = 0; dt < 3; ++dt)
      launch_gemm(stream, Gb, 64, 1, dt, 0, twT + (size_t)dt * 4096, 64, 0,
                  OUTb, 64, 0, nullptr, 0, 1, NN_ * TT, 64, 64, 1);
    // relu + LayerNorm over (N,O) per t, write (B,N,O,T)
    k_stats<<<TT, 256, 0, stream>>>(OUTb, stats);
    k_final<<<(NN_ * OO * TT + 255) / 256, 256, 0, stream>>>(
        OUTb, stats, lnw, lnb, out + (size_t)b * NN_ * OO * TT);
  }
}
